// AttnBlock_19859928777254
// MI455X (gfx1250) — compile-verified, matrix operand order checked by reading, not run on hardware
//
#include <hip/hip_runtime.h>

typedef __bf16 bf16;
typedef __attribute__((ext_vector_type(16))) __bf16 v16bf;
typedef __attribute__((ext_vector_type(8)))  float  v8f;

#define N_SP   4096
#define C_CH   128
#define BATCH  4
#define NGROUP 32

// K-index packing for 16-bit A/B fragments (ISA 7.12.2):
// VGPR v holds K pair {kpair, kpair+1}; lanes 0-15 vs 16-31 differ by +8.
static __device__ __forceinline__ int kpair(int v, int half) {
    return (v < 4 ? 2 * v : 8 + 2 * v) + 8 * half;
}

static __device__ __forceinline__ v8f wmma_bf16(v16bf a, v16bf b, v8f c) {
    return __builtin_amdgcn_wmma_f32_16x16x32_bf16(
        /*neg_a=*/false, a, /*neg_b=*/false, b,
        /*c_mod=*/(short)0, c, /*reuse_a=*/false, /*reuse_b=*/false);
}

typedef union { v16bf v; uint u[8]; } fragU;

// A-style fragment: 16x32 tile from row-major [*, ld] bf16, rows row0.., cols k0..
// K pairs are contiguous -> 8 aligned dword loads, no b16 repacking.
// Also used for B operands stored TRANSPOSED (B^T row-major == A layout).
static __device__ __forceinline__ v16bf load_a_g(const bf16* __restrict__ base,
                                                 int ld, int row0, int k0, int lane) {
    int half = lane >> 4, m = lane & 15;
    const bf16* r = base + (size_t)(row0 + m) * ld + k0;
    fragU f;
#pragma unroll
    for (int v = 0; v < 8; ++v)
        f.u[v] = *(const uint*)(r + kpair(v, half));
    return f.v;
}

// ---------------- GroupNorm statistics: one block per (batch, group) ----------------
__global__ void gn_stats_kernel(const float* __restrict__ x, float* __restrict__ stats) {
    const int GEL = (C_CH / NGROUP) * N_SP;  // 16384 contiguous floats per group
    int b = blockIdx.x >> 5, g = blockIdx.x & 31, tid = threadIdx.x;
    const float* base = x + ((size_t)b * C_CH + g * (C_CH / NGROUP)) * N_SP;
    float s = 0.f, ss = 0.f;
    for (int i = tid; i < GEL; i += 256) {
        float v = base[i];
        s += v; ss += v * v;
    }
    __shared__ float rs[256], rss[256];
    rs[tid] = s; rss[tid] = ss;
    __syncthreads();
    for (int off = 128; off > 0; off >>= 1) {
        if (tid < off) { rs[tid] += rs[tid + off]; rss[tid] += rss[tid + off]; }
        __syncthreads();
    }
    if (tid == 0) {
        float m   = rs[0] / (float)GEL;
        float var = rss[0] / (float)GEL - m * m;
        stats[blockIdx.x * 2]     = m;
        stats[blockIdx.x * 2 + 1] = rsqrtf(var + 1e-5f);
    }
}

// ---------------- Normalize + affine, write channels-last bf16 h[B,N,C] ----------------
__global__ void gn_apply_kernel(const float* __restrict__ x, const float* __restrict__ w,
                                const float* __restrict__ bb, const float* __restrict__ stats,
                                bf16* __restrict__ hbf) {
    int idx = blockIdx.x * 256 + threadIdx.x;           // over B*C*N = 2^21
    int n = idx & (N_SP - 1);
    int c = (idx >> 12) & (C_CH - 1);
    int b = idx >> 19;
    int g = c >> 2;
    float mean = stats[(b * NGROUP + g) * 2];
    float rstd = stats[(b * NGROUP + g) * 2 + 1];
    float v = (x[idx] - mean) * rstd * w[c] + bb[c];
    hbf[((size_t)b * N_SP + n) * C_CH + c] = (bf16)v;
}

// ---------------- Weights fp32 -> bf16, TRANSPOSED (Wt[d][c]), Wq|Wk|Wv|Wo ----------------
__global__ void wconv_kernel(const float* __restrict__ wq, const float* __restrict__ wk,
                             const float* __restrict__ wv, const float* __restrict__ wo,
                             bf16* __restrict__ dst) {
    int idx = blockIdx.x * 256 + threadIdx.x;   // 4 * 16384 outputs
    int which = idx >> 14, i = idx & 16383;
    int d = i >> 7, c = i & 127;                // output index (d,c) <- src (c,d)
    const float* src = which == 0 ? wq : which == 1 ? wk : which == 2 ? wv : wo;
    dst[idx] = (bf16)src[c * C_CH + d];
}

// ---------------- QKV projection: one wave per 16-row tile ----------------
__global__ void qkv_kernel(const bf16* __restrict__ hbf, const bf16* __restrict__ wqkvT,
                           const float* __restrict__ bq, const float* __restrict__ bk,
                           const float* __restrict__ bv,
                           bf16* __restrict__ qbf, bf16* __restrict__ kbf, bf16* __restrict__ vbf) {
    int tid = threadIdx.x, lane = tid & 31, wave = tid >> 5;
    int half = lane >> 4, nn = lane & 15;
    int row0 = blockIdx.x * 128 + wave * 16;

    v16bf afr[4];
#pragma unroll
    for (int ks = 0; ks < 4; ++ks) afr[ks] = load_a_g(hbf, C_CH, row0, ks * 32, lane);

    const float* biases[3] = {bq, bk, bv};
    bf16* dsts[3] = {qbf, kbf, vbf};
    const float qscale = 0.08838834764831845f;  // C^-0.5, folded into Q

#pragma unroll
    for (int mtx = 0; mtx < 3; ++mtx) {
        const bf16* WT = wqkvT + mtx * C_CH * C_CH;   // transposed: WT[d][c]
        float scale = (mtx == 0) ? qscale : 1.0f;
        for (int d0 = 0; d0 < C_CH; d0 += 16) {
            v8f acc = {};
#pragma unroll
            for (int ks = 0; ks < 4; ++ks)
                acc = wmma_bf16(afr[ks], load_a_g(WT, C_CH, d0, ks * 32, lane), acc);
            float bias = biases[mtx][d0 + nn];
#pragma unroll
            for (int v = 0; v < 8; ++v) {
                int row = row0 + v + 8 * half;
                dsts[mtx][(size_t)row * C_CH + d0 + nn] = (bf16)((acc[v] + bias) * scale);
            }
        }
    }
}

// ---------------- Flash attention: 8 waves share LDS K/V tiles ----------------
__global__ void attn_kernel(const bf16* __restrict__ qbf, const bf16* __restrict__ kbf,
                            const bf16* __restrict__ vbf, bf16* __restrict__ obf) {
    __shared__ bf16 Kl[32 * C_CH];       // K tile  [32 keys][128 ch]  (row major)
    __shared__ bf16 Vt[C_CH * 32];       // V tile  [128 ch][32 keys]  (TRANSPOSED)
    __shared__ bf16 Pl[8][16 * 32];      // per-wave P staging (C/D -> A re-layout)

    int tid = threadIdx.x, lane = tid & 31, wave = tid >> 5;
    int half = lane >> 4, nn = lane & 15;
    int b  = blockIdx.x >> 5;
    int q0 = (blockIdx.x & 31) * 128 + wave * 16;

    const bf16* qb = qbf + (size_t)b * N_SP * C_CH;
    const bf16* kb = kbf + (size_t)b * N_SP * C_CH;
    const bf16* vb = vbf + (size_t)b * N_SP * C_CH;

    v16bf qfr[4];
#pragma unroll
    for (int ks = 0; ks < 4; ++ks) qfr[ks] = load_a_g(qb, C_CH, q0, ks * 32, lane);

    v8f oacc[8];
    v8f zero = {};
#pragma unroll
    for (int d = 0; d < 8; ++d) oacc[d] = zero;
    float rowmax[8], rowsum[8];
#pragma unroll
    for (int v = 0; v < 8; ++v) { rowmax[v] = -3.0e38f; rowsum[v] = 0.f; }

    uint* Klw = (uint*)Kl;

    for (int kt = 0; kt < N_SP / 32; ++kt) {
        // K tile: straight coalesced copy (2048 dwords, 8 per thread)
        const uint* ksrc = (const uint*)(kb + (size_t)kt * 32 * C_CH);
        const uint* vsrc = (const uint*)(vb + (size_t)kt * 32 * C_CH);
#pragma unroll
        for (int i = 0; i < 8; ++i)
            Klw[tid + i * 256] = ksrc[tid + i * 256];
        // V tile: coalesced dword read, transpose into Vt[ch][key]
#pragma unroll
        for (int i = 0; i < 8; ++i) {
            int w = tid + i * 256;                 // dword index in 32x128 tile
            int key = w >> 6, ch = (w & 63) * 2;   // dword = channels (ch, ch+1)
            union { uint u; bf16 h[2]; } t;
            t.u = vsrc[w];
            Vt[ch * 32 + key]       = t.h[0];
            Vt[(ch + 1) * 32 + key] = t.h[1];
        }
        if (kt + 1 < N_SP / 32) {  // global_prefetch_b8 of next tile
            __builtin_prefetch(kb + (size_t)(kt + 1) * 32 * C_CH + tid * 32, 0, 1);
            __builtin_prefetch(vb + (size_t)(kt + 1) * 32 * C_CH + tid * 32, 0, 1);
        }
        __syncthreads();

        // S = Q (16x128) * K^T -> two 16x16 tiles (keys j*16..j*16+15).
        // B = K^T, and K tile is row-major [key][ch] == (K^T)^T -> A-style load.
        v8f sacc[2];
#pragma unroll
        for (int j = 0; j < 2; ++j) {
            v8f a = zero;
#pragma unroll
            for (int ks = 0; ks < 4; ++ks)
                a = wmma_bf16(qfr[ks], load_a_g(Kl, C_CH, j * 16, ks * 32, lane), a);
            sacc[j] = a;
        }

        // online softmax; rows live in VGPR v (row = v + 8*half), 16-lane groups
        float p0[8], p1[8], corr[8];
#pragma unroll
        for (int v = 0; v < 8; ++v) {
            float m = fmaxf(sacc[0][v], sacc[1][v]);
#pragma unroll
            for (int off = 8; off > 0; off >>= 1)
                m = fmaxf(m, __shfl_xor(m, off, 32));
            float nm = fmaxf(rowmax[v], m);
            corr[v]  = __expf(rowmax[v] - nm);
            rowmax[v] = nm;
            p0[v] = __expf(sacc[0][v] - nm);
            p1[v] = __expf(sacc[1][v] - nm);
            float s = p0[v] + p1[v];
#pragma unroll
            for (int off = 8; off > 0; off >>= 1)
                s += __shfl_xor(s, off, 32);
            rowsum[v] = rowsum[v] * corr[v] + s;
        }
#pragma unroll
        for (int d = 0; d < 8; ++d)
#pragma unroll
            for (int v = 0; v < 8; ++v) oacc[d][v] *= corr[v];

        // re-layout P (C/D format) into A format via wave-private LDS patch
        bf16* P = Pl[wave];
#pragma unroll
        for (int v = 0; v < 8; ++v) {
            int m = v + 8 * half;
            P[m * 32 + nn]      = (bf16)p0[v];
            P[m * 32 + 16 + nn] = (bf16)p1[v];
        }
        v16bf pa = load_a_g(P, 32, 0, 0, lane);

        // O += P (16x32) * V (32x128); B = V, Vt is V^T -> A-style load
#pragma unroll
        for (int d = 0; d < 8; ++d)
            oacc[d] = wmma_bf16(pa, load_a_g(Vt, 32, d * 16, 0, lane), oacc[d]);
        __syncthreads();
    }

#pragma unroll
    for (int v = 0; v < 8; ++v) rowsum[v] = 1.0f / rowsum[v];
    bf16* ob = obf + (size_t)b * N_SP * C_CH;
#pragma unroll
    for (int d = 0; d < 8; ++d)
#pragma unroll
        for (int v = 0; v < 8; ++v) {
            int row = q0 + v + 8 * half;
            ob[(size_t)row * C_CH + d * 16 + nn] = (bf16)(oacc[d][v] * rowsum[v]);
        }
}

// ---------------- Output projection + residual, back to [B,C,H,W] ----------------
__global__ void proj_kernel(const bf16* __restrict__ obf, const bf16* __restrict__ woT,
                            const float* __restrict__ bo, const float* __restrict__ x,
                            float* __restrict__ out) {
    int tid = threadIdx.x, lane = tid & 31, wave = tid >> 5;
    int half = lane >> 4, nn = lane & 15;
    int row0 = blockIdx.x * 128 + wave * 16;

    v16bf afr[4];
#pragma unroll
    for (int ks = 0; ks < 4; ++ks) afr[ks] = load_a_g(obf, C_CH, row0, ks * 32, lane);

    for (int d0 = 0; d0 < C_CH; d0 += 16) {
        v8f acc = {};
#pragma unroll
        for (int ks = 0; ks < 4; ++ks)
            acc = wmma_bf16(afr[ks], load_a_g(woT, C_CH, d0, ks * 32, lane), acc);
        float bias = bo[d0 + nn];
#pragma unroll
        for (int v = 0; v < 8; ++v) {
            int pos = row0 + v + 8 * half;        // flat token index over B*N
            int bb = pos >> 12, nsp = pos & (N_SP - 1);
            size_t oi = ((size_t)bb * C_CH + d0 + nn) * N_SP + nsp;
            out[oi] = x[oi] + acc[v] + bias;
        }
    }
}

extern "C" void kernel_launch(void* const* d_in, const int* in_sizes, int n_in,
                              void* d_out, int out_size, void* d_ws, size_t ws_size,
                              hipStream_t stream) {
    (void)in_sizes; (void)n_in; (void)out_size; (void)ws_size;
    const float* x  = (const float*)d_in[0];
    const float* nw = (const float*)d_in[1];
    const float* nb = (const float*)d_in[2];
    const float* Wq = (const float*)d_in[3];
    const float* bq = (const float*)d_in[4];
    const float* Wk = (const float*)d_in[5];
    const float* bk = (const float*)d_in[6];
    const float* Wv = (const float*)d_in[7];
    const float* bv = (const float*)d_in[8];
    const float* Wo = (const float*)d_in[9];
    const float* bo = (const float*)d_in[10];
    float* out = (float*)d_out;

    // workspace layout (256B aligned)
    char* ws = (char*)d_ws;
    float* stats = (float*)ws;                       //   1 KB: mean/rstd per (b,g)
    bf16* wbf = (bf16*)(ws + 1024);                  // 128 KB: Wq|Wk|Wv|Wo bf16 (transposed)
    bf16* hbf = (bf16*)(ws + 132096);                //   4 MB: normalized h [B,N,C]
    bf16* qbf = (bf16*)(ws + 4326400);               //   4 MB
    bf16* kbf = (bf16*)(ws + 8520704);               //   4 MB
    bf16* vbf = (bf16*)(ws + 12715008);              //   4 MB
    bf16* obf = (bf16*)(ws + 16909312);              //   4 MB: attention output

    gn_stats_kernel<<<BATCH * NGROUP, 256, 0, stream>>>(x, stats);
    gn_apply_kernel<<<(BATCH * C_CH * N_SP) / 256, 256, 0, stream>>>(x, nw, nb, stats, hbf);
    wconv_kernel<<<(4 * C_CH * C_CH) / 256, 256, 0, stream>>>(Wq, Wk, Wv, Wo, wbf);
    qkv_kernel<<<(BATCH * N_SP) / 128, 256, 0, stream>>>(hbf, wbf, bq, bk, bv, qbf, kbf, vbf);
    attn_kernel<<<BATCH * (N_SP / 128), 256, 0, stream>>>(qbf, kbf, vbf, obf);
    proj_kernel<<<(BATCH * N_SP) / 128, 256, 0, stream>>>(obf, wbf + 3 * C_CH * C_CH, bo, x, out);
}
